// QLayer_47407849013332
// MI455X (gfx1250) — matrix operand order repaired and checked by reading, not compiled, not run on hardware
//
#include <hip/hip_runtime.h>
#include <hip/hip_bf16.h>

typedef __attribute__((ext_vector_type(16))) __bf16 v16bf;
typedef __attribute__((ext_vector_type(8)))  float  v8f;
typedef __attribute__((ext_vector_type(4)))  unsigned int v4u;
typedef __attribute__((ext_vector_type(8)))  int v8i;
typedef __attribute__((ext_vector_type(4)))  int v4i;

#define S_ 8
#define B_ 8
#define T_ 2048
#define D_ 64
#define K_ 512
#define ROWS (B_*T_)      /* 16384 */
#define RPW 16            /* rows per wave tile */
#define WAVES 8
#define RPB (RPW*WAVES)   /* 128 rows per block */
#define NCB (K_/16)       /* 32 code blocks */

// workspace layout (bytes, 512-aligned)
#define WS_EBF  0              /* S*K*D bf16 = 524288 */
#define WS_WT   524288         /* S*D*D bf16 (transposed [s][e][d]) = 65536 */
#define WS_E2   589824         /* S*K f32 = 16384 */
#define WS_BNA  606208         /* S*D f32 */
#define WS_BNB  608256         /* S*D f32 */

#if __has_builtin(__builtin_amdgcn_tensor_load_to_lds)
#define HAVE_TDM 1
#endif

// 2KB contiguous global->LDS async copy: 32 lanes x 16B x 4 (ASYNCcnt +4).
// INST_OFFSET advances both LDS and global addresses (ISA 10.7/15.18 async pseudocode).
__device__ __forceinline__ void async_copy_2k(unsigned ldst, unsigned long long gsrc) {
  asm volatile(
      "global_load_async_to_lds_b128 %0, %1, off\n\t"
      "global_load_async_to_lds_b128 %0, %1, off offset:512\n\t"
      "global_load_async_to_lds_b128 %0, %1, off offset:1024\n\t"
      "global_load_async_to_lds_b128 %0, %1, off offset:1536"
      :: "v"(ldst), "v"(gsrc) : "memory");
}
#define ASYNC_WAIT(n) asm volatile("s_wait_asynccnt " #n ::: "memory")

__device__ __forceinline__ unsigned lds_off(const void* p) {
  return (unsigned)(unsigned long long)p;   // low 32 bits of generic ptr = LDS byte address
}

__global__ void prep_w(const float* __restrict__ W, const float* __restrict__ gamma,
                       const float* __restrict__ beta, const float* __restrict__ mean,
                       const float* __restrict__ var, const float* __restrict__ bias,
                       __bf16* __restrict__ Wt, float* __restrict__ bnA, float* __restrict__ bnB) {
  int i = blockIdx.x * blockDim.x + threadIdx.x;
  if (i < S_*D_*D_) {
    int s = i / (D_*D_); int r = i % (D_*D_); int e = r / D_; int d = r % D_;
    Wt[i] = (__bf16)W[(s*D_ + d)*D_ + e];
  }
  if (i < S_*D_) {
    float inv = rsqrtf(var[i] + 1e-3f);
    float A = gamma[i] * inv;
    bnA[i] = A;
    bnB[i] = beta[i] + A * (bias[i] - mean[i]);
  }
}

__global__ void prep_e(const float* __restrict__ E, __bf16* __restrict__ Ebf,
                       float* __restrict__ e2) {
  int i = blockIdx.x * blockDim.x + threadIdx.x;
  if (i < S_*K_*D_) Ebf[i] = (__bf16)E[i];
  if (i < S_*K_) {
    const float* er = E + (size_t)i * D_;
    float acc = 0.f;
    #pragma unroll
    for (int d = 0; d < D_; ++d) acc += er[d] * er[d];
    e2[i] = acc;
  }
}

__global__ __launch_bounds__(256) void vq_fused(
    const float* __restrict__ x, const float* __restrict__ E,
    const __bf16* __restrict__ Wt, const __bf16* __restrict__ Ebf,
    const float* __restrict__ e2, const float* __restrict__ bnA,
    const float* __restrict__ bnB,
    float* __restrict__ zq, float* __restrict__ ze, long long* __restrict__ zk)
{
  __shared__ float  xs[RPB * D_];             // 32 KB fp32 x tile
  __shared__ __bf16 zsm[WAVES][RPW * D_];     // 16 KB per-wave z staging
  __shared__ __bf16 ebuf[WAVES][2][16 * D_];  // 32 KB per-wave E double buffer
  __shared__ float  e2s[WAVES][K_];           // 16 KB per-wave ||E||^2

  const int tid  = threadIdx.x;
  const int wave = tid >> 5, lane = tid & 31;
  const int hl   = lane >> 4, l16 = lane & 15;
  const size_t rowblk = (size_t)blockIdx.x * RPB;

  // ---------- x tile: 128 rows x 64 f32, contiguous 32KB ----------
#ifdef HAVE_TDM
  if (wave == 0) {
    unsigned long long ga = (unsigned long long)(x + rowblk * D_);
    v4u g0;
    g0[0] = 1u;                                   // count=1, user descriptor
    g0[1] = lds_off(xs);                          // lds_addr
    g0[2] = (unsigned)(ga & 0xffffffffu);         // global_addr[31:0]
    g0[3] = (unsigned)((ga >> 32) & 0x01ffffffu) | 0x80000000u;  // addr[56:32] | type=2
    v8i g1;
    g1[0] = 0x00020000;          // workgroup_mask=0, data_size=2 (4B)
    g1[1] = (int)(64u << 16);    // tensor_dim0 = 64 (bits 79:48)
    g1[2] = (int)(16384u << 16); // tensor_dim1 = 16384 (bits 111:80)
    g1[3] = (int)(64u << 16);    // tile_dim0 = 64 (bits 127:112)
    g1[4] = (int)RPB;            // tile_dim1 = 128, tile_dim2 = 0
    g1[5] = (int)64;             // tensor_dim0_stride = 64 (bits 207:160)
    g1[6] = 0;
    g1[7] = 0;
    v4i g2 = (v4i){0,0,0,0};
    v4i g3 = (v4i){0,0,0,0};
#if __clang_major__ >= 23
    v8i g4 = (v8i){0,0,0,0,0,0,0,0};
    __builtin_amdgcn_tensor_load_to_lds(g0, g1, g2, g3, g4, 0);
#else
    __builtin_amdgcn_tensor_load_to_lds(g0, g1, g2, g3, 0);
#endif
    __builtin_amdgcn_s_wait_tensorcnt(0);
  }
#else
  {
    const float4* xg = (const float4*)(x + rowblk * D_);
    float4* xl = (float4*)xs;
    for (int i = tid; i < RPB * D_ / 4; i += 256) xl[i] = xg[i];
  }
#endif
  __syncthreads();

  // x A-fragments (16x32 bf16 A layout), reused for all 8 scales
  const float* xrow = &xs[(wave * RPW + l16) * D_];
  v16bf xa[2];
  #pragma unroll
  for (int j = 0; j < 2; ++j) {
    int kb = 32 * j + hl * 16;
    #pragma unroll
    for (int q = 0; q < 8; ++q) {
      xa[j][q]     = (__bf16)xrow[kb + q];
      xa[j][8 + q] = (__bf16)xrow[kb + 8 + q];
    }
  }

  v8f zeacc[4], zqacc[4];
  #pragma unroll
  for (int t = 0; t < 4; ++t) { zeacc[t] = (v8f){}; zqacc[t] = (v8f){}; }

  __bf16* zsl = zsm[wave];
  const unsigned ebuf_l = lds_off(&ebuf[wave][0][0]) + (unsigned)(lane * 16);
  const unsigned e2s_l  = lds_off(&e2s[wave][0])     + (unsigned)(lane * 16);

  for (int s = 0; s < S_; ++s) {
    // ---------- projection: proj(16x64) = x_tile @ W[s] ----------
    v8f proj[4];
    #pragma unroll
    for (int t = 0; t < 4; ++t) {
      v8f acc = (v8f){};
      #pragma unroll
      for (int j = 0; j < 2; ++j) {
        const v16bf bfrag = *(const v16bf*)(Wt +
            ((size_t)(s * D_ + (t * 16 + l16)) * D_ + 32 * j + hl * 16));
        acc = __builtin_amdgcn_wmma_f32_16x16x32_bf16(false, xa[j], false, bfrag,
                                                      (short)0, acc, false, false);
      }
      proj[t] = acc;
    }

    // kick off async staging: ||E||^2 row block (2KB) then first code block (2KB)
    async_copy_2k(e2s_l,  (unsigned long long)(e2 + s * K_) + (unsigned)(lane * 16));
    async_copy_2k(ebuf_l, (unsigned long long)(Ebf + (size_t)s * K_ * D_) + (unsigned)(lane * 16));

    // ---------- fused BN affine ----------
    #pragma unroll
    for (int t = 0; t < 4; ++t) {
      float A  = bnA[s * D_ + t * 16 + l16];
      float Bc = bnB[s * D_ + t * 16 + l16];
      #pragma unroll
      for (int r = 0; r < 8; ++r) proj[t][r] = proj[t][r] * A + Bc;
    }

    // ---------- softmax over the 64 features of each row ----------
    float rinv[8];
    #pragma unroll
    for (int r = 0; r < 8; ++r) {
      float m = fmaxf(fmaxf(proj[0][r], proj[1][r]), fmaxf(proj[2][r], proj[3][r]));
      #pragma unroll
      for (int msk = 1; msk < 16; msk <<= 1) m = fmaxf(m, __shfl_xor(m, msk, 32));
      float sum = 0.f;
      #pragma unroll
      for (int t = 0; t < 4; ++t) { float ev = __expf(proj[t][r] - m); proj[t][r] = ev; sum += ev; }
      #pragma unroll
      for (int msk = 1; msk < 16; msk <<= 1) sum += __shfl_xor(sum, msk, 32);
      rinv[r] = 1.0f / sum;
    }

    // ---------- z = x * att ; accumulate z_e ; z2 row norms ; stage z (bf16) ----------
    __syncthreads();   // WAR protection on zsl across s-iterations (uniform)
    float z2[8];
    #pragma unroll
    for (int r = 0; r < 8; ++r) {
      int M = r + hl * 8;
      float part = 0.f;
      #pragma unroll
      for (int t = 0; t < 4; ++t) {
        float xv = xs[(wave * RPW + M) * D_ + t * 16 + l16];
        float zv = proj[t][r] * rinv[r] * xv;
        zeacc[t][r] += zv;
        part += zv * zv;
        zsl[M * D_ + t * 16 + l16] = (__bf16)zv;
      }
      #pragma unroll
      for (int msk = 1; msk < 16; msk <<= 1) part += __shfl_xor(part, msk, 32);
      z2[r] = part;
    }
    __syncthreads();

    // ---------- z A-fragments ----------
    v16bf za[2];
    const __bf16* zrow = &zsl[l16 * D_];
    #pragma unroll
    for (int j = 0; j < 2; ++j) {
      int kb = 32 * j + hl * 16;
      #pragma unroll
      for (int q = 0; q < 8; ++q) { za[j][q] = zrow[kb + q]; za[j][8 + q] = zrow[kb + 8 + q]; }
    }

    // ---------- distance GEMM: async double-buffered E blocks ----------
    float minv[8]; int mini[8];
    #pragma unroll
    for (int r = 0; r < 8; ++r) { minv[r] = 3.4e38f; mini[r] = 0; }
    const unsigned long long egs = (unsigned long long)(Ebf + (size_t)s * K_ * D_)
                                 + (unsigned)(lane * 16);

    for (int cb = 0; cb < NCB; ++cb) {
      const int buf = cb & 1;
      if (cb + 1 < NCB) {
        async_copy_2k(ebuf_l + (unsigned)((buf ^ 1) * 2048),
                      egs + (unsigned long long)(cb + 1) * 2048);
        ASYNC_WAIT(4);   // retire current block (+ e2 rows on first iteration)
      } else {
        ASYNC_WAIT(0);
      }
      const __bf16* eb = &ebuf[wave][buf][0];
      v8f acc = (v8f){};
      #pragma unroll
      for (int j = 0; j < 2; ++j) {
        const v16bf bfrag = *(const v16bf*)(eb + (size_t)l16 * D_ + 32 * j + hl * 16);
        acc = __builtin_amdgcn_wmma_f32_16x16x32_bf16(false, za[j], false, bfrag,
                                                      (short)0, acc, false, false);
      }
      float e2v = e2s[wave][cb * 16 + l16];
      int idx = cb * 16 + l16;
      #pragma unroll
      for (int r = 0; r < 8; ++r) {
        float d2 = z2[r] + e2v - 2.0f * acc[r];
        if (d2 < minv[r]) { minv[r] = d2; mini[r] = idx; }
      }
    }

    // cross-lane argmin (within each 16-lane half), first-index wins on ties
    #pragma unroll
    for (int r = 0; r < 8; ++r) {
      #pragma unroll
      for (int msk = 1; msk < 16; msk <<= 1) {
        float ov = __shfl_xor(minv[r], msk, 32);
        int   oi = __shfl_xor(mini[r], msk, 32);
        if (ov < minv[r] || (ov == minv[r] && oi < mini[r])) { minv[r] = ov; mini[r] = oi; }
      }
    }

    // ---------- write z_k, gather winning E rows (fp32) into z_q accumulator ----------
    const size_t rowg0 = rowblk + (size_t)wave * RPW;
    if (l16 == 0) {
      #pragma unroll
      for (int r = 0; r < 8; ++r) {
        int M = r + hl * 8;
        zk[(size_t)s * ROWS + rowg0 + M] = (long long)mini[r];
      }
    }
    const float* Es = E + (size_t)s * K_ * D_;
    #pragma unroll
    for (int r = 0; r < 8; ++r) {
      const float* erow = Es + (size_t)mini[r] * D_;
      #pragma unroll
      for (int t = 0; t < 4; ++t) zqacc[t][r] += erow[t * 16 + l16];
    }
  }

  // ---------- write z_e / z_q ----------
  const size_t rowg0 = rowblk + (size_t)wave * RPW;
  #pragma unroll
  for (int r = 0; r < 8; ++r) {
    int M = r + hl * 8;
    size_t base = (rowg0 + M) * D_;
    #pragma unroll
    for (int t = 0; t < 4; ++t) {
      ze[base + t * 16 + l16] = zeacc[t][r];
      zq[base + t * 16 + l16] = zqacc[t][r];
    }
  }
}

extern "C" void kernel_launch(void* const* d_in, const int* in_sizes, int n_in,
                              void* d_out, int out_size, void* d_ws, size_t ws_size,
                              hipStream_t stream) {
  const float* x       = (const float*)d_in[0];
  const float* W       = (const float*)d_in[1];
  const float* b       = (const float*)d_in[2];
  const float* gamma   = (const float*)d_in[3];
  const float* beta    = (const float*)d_in[4];
  const float* bn_mean = (const float*)d_in[5];
  const float* bn_var  = (const float*)d_in[6];
  const float* E       = (const float*)d_in[7];

  char* ws = (char*)d_ws;
  __bf16* Ebf = (__bf16*)(ws + WS_EBF);
  __bf16* Wt  = (__bf16*)(ws + WS_WT);
  float*  e2  = (float*)(ws + WS_E2);
  float*  bnA = (float*)(ws + WS_BNA);
  float*  bnB = (float*)(ws + WS_BNB);

  float* zq = (float*)d_out;
  float* ze = zq + (size_t)B_ * T_ * D_;
  long long* zk = (long long*)(ze + (size_t)B_ * T_ * D_);

  prep_w<<<(S_*D_*D_ + 255) / 256, 256, 0, stream>>>(W, gamma, beta, bn_mean, bn_var, b,
                                                     Wt, bnA, bnB);
  prep_e<<<(S_*K_*D_ + 255) / 256, 256, 0, stream>>>(E, Ebf, e2);
  vq_fused<<<ROWS / RPB, 256, 0, stream>>>(x, E, Wt, Ebf, e2, bnA, bnB, zq, ze, zk);
}
